// scaledDotProductAttention_75265006895186
// MI455X (gfx1250) — matrix operand order repaired and checked by reading, not compiled, not run on hardware
//
#include <hip/hip_runtime.h>

typedef _Float16 v16h __attribute__((ext_vector_type(16)));
typedef _Float16 v8h  __attribute__((ext_vector_type(8)));
typedef float    v8f  __attribute__((ext_vector_type(8)));
typedef float    v4f  __attribute__((ext_vector_type(4)));

// Problem constants (match reference)
constexpr int Bc = 16;
constexpr int Nc = 2048;
constexpr int Mc = 2048;
constexpr int Dc = 128;   // head dim (QK contraction)
constexpr int Vc = 128;   // value dim
constexpr int QT = 128;   // query rows per block (8 waves x 16)
constexpr int KB = 32;    // key block (WMMA K for PV gemm)

// softmax in log2 domain: fold log2(e) into the score scale
constexpr float kScale2 = 0.12751649737243868f;      // (1/sqrt(128)) * log2(e)
constexpr float kMask2  = -1.4426950408889634e6f;    // -1e6 * log2(e)

static __device__ inline float fexp2(float x) {
#if __has_builtin(__builtin_amdgcn_exp2f)
    return __builtin_amdgcn_exp2f(x);                // raw v_exp_f32
#else
    return __expf(x * 0.6931471805599453f);
#endif
}

__global__ __launch_bounds__(256, 1)
void fa2_wmma_gfx1250(const float* __restrict__ Q,
                      const float* __restrict__ K,
                      const float* __restrict__ Vv,
                      const int*   __restrict__ kpl,
                      const int*   __restrict__ attnflag,
                      float*       __restrict__ Out)
{
    // Double-buffered LDS staging: K row-major (keys x D), V transposed (vcol x keys).
    // Row pads give a 4-bank shift per row -> conflict-free b128 reads across 16 rows.
    __shared__ _Float16 ldsK [2][KB][Dc + 8];
    __shared__ _Float16 ldsVt[2][Vc][KB + 8];

    const int tid = threadIdx.x;
    const int wv  = tid >> 5;        // wave id 0..7
    const int ln  = tid & 31;
    const int lh  = ln & 15;         // lane % 16
    const int hb  = ln >> 4;         // 0 = lanes 0-15, 1 = lanes 16-31

    const int b    = blockIdx.x / (Nc / QT);
    const int qt   = (blockIdx.x % (Nc / QT)) * QT;
    const int qrow = qt + wv * 16 + lh;      // query row this lane represents

    const int causal  = attnflag[0];
    const int keepEnd = Mc - kpl[b];         // keys >= keepEnd are zeroed (reference mask())

    // Per-thread staging geometry (constant across iterations)
    int srow[4], scol[4];
    #pragma unroll
    for (int j = 0; j < 4; ++j) {
        const int f4 = tid + 256 * j;
        srow[j] = f4 >> 5;
        scol[j] = f4 & 31;
    }

    // ---- Q as B-operand of S^T = K * Q^T : lane q holds Q[qrow][d], d = 32f+8*hb+{0..7,16..23} ----
    v16h bq[4];
    {
        const float* qp = Q + ((size_t)b * Nc + qrow) * Dc;
        #pragma unroll
        for (int f = 0; f < 4; ++f) {
            const float* p = qp + 32 * f + 8 * hb;
            #pragma unroll
            for (int i = 0; i < 16; ++i) bq[f][i] = (_Float16)p[i < 8 ? i : i + 8];
        }
    }

    // ---- O^T accumulators: 8 tiles of 16 vdims x 16 queries ----
    v8f o[8];
    #pragma unroll
    for (int t = 0; t < 8; ++t)
        #pragma unroll
        for (int r = 0; r < 8; ++r) o[t][r] = 0.0f;

    float m = -3.0e38f;   // running row max (log2 domain, replicated lanes q / q+16)
    float l = 0.0f;       // running row sum

    const int kbEnd = causal ? (qt + QT) : Mc;
    const int nblk  = kbEnd / KB;

    // Wave-uniform causal structure:
    //  i <  numFull      : block fully below diagonal -> no masking needed
    //  i >= computeEnd   : block fully above diagonal for this wave -> skip compute
    const int numFull    = causal ? ((qt + 16 * wv + 1) >> 5) : nblk;
    const int computeEnd = causal ? (((qt + 16 * wv + 15) >> 5) + 1) : nblk;

    // ---- register-staged double buffering ----
    float4 kreg[4], vreg[4];

    auto loadBlk = [&](int kb) {
        #pragma unroll
        for (int j = 0; j < 4; ++j) {
            kreg[j] = *((const float4*)(K  + ((size_t)b * Mc + kb + srow[j]) * Dc) + scol[j]);
            vreg[j] = *((const float4*)(Vv + ((size_t)b * Mc + kb + srow[j]) * Vc) + scol[j]);
            if (kb + srow[j] >= keepEnd) {        // zero padded key rows (reference mask())
                kreg[j].x = 0.f; kreg[j].y = 0.f; kreg[j].z = 0.f; kreg[j].w = 0.f;
            }
        }
    };
    auto storeBlk = [&](int buf) {
        #pragma unroll
        for (int j = 0; j < 4; ++j) {
            _Float16* dk = &ldsK[buf][srow[j]][scol[j] * 4];
            dk[0] = (_Float16)kreg[j].x; dk[1] = (_Float16)kreg[j].y;
            dk[2] = (_Float16)kreg[j].z; dk[3] = (_Float16)kreg[j].w;
            ldsVt[buf][scol[j] * 4 + 0][srow[j]] = (_Float16)vreg[j].x;
            ldsVt[buf][scol[j] * 4 + 1][srow[j]] = (_Float16)vreg[j].y;
            ldsVt[buf][scol[j] * 4 + 2][srow[j]] = (_Float16)vreg[j].z;
            ldsVt[buf][scol[j] * 4 + 3][srow[j]] = (_Float16)vreg[j].w;
        }
    };

    // Prologue: block 0 into buffer 0
    loadBlk(0);
    storeBlk(0);

    for (int i = 0; i < nblk; ++i) {
        const int kb  = i * KB;
        const int buf = i & 1;

        // A: issue global loads for block i+1 (latency hidden by compute below)
        if (i + 1 < nblk) {
            loadBlk(kb + KB);
            #pragma unroll
            for (int j = 0; j < 4; ++j) {
                if (i + 2 < nblk) {
                    __builtin_prefetch(K  + ((size_t)b * Mc + kb + 2 * KB + srow[j]) * Dc + scol[j] * 4, 0, 0);
                    __builtin_prefetch(Vv + ((size_t)b * Mc + kb + 2 * KB + srow[j]) * Vc + scol[j] * 4, 0, 0);
                }
            }
        }

        // B: single barrier per iteration — publishes lds[buf], protects lds[buf^1] rewrite
        __syncthreads();

        // Fully-masked block for this wave: contributes exactly zero — skip compute.
        if (i < computeEnd) {
            // C: ---- GEMM 1: S^T = K_tile * Q^T, two 16-key subtiles ----
            v8f c1, c2;
            #pragma unroll
            for (int r = 0; r < 8; ++r) { c1[r] = 0.f; c2[r] = 0.f; }
            #pragma unroll
            for (int f = 0; f < 4; ++f) {
                v16h a0, a1;
                {
                    const _Float16* r0 = &ldsK[buf][lh][32 * f + 8 * hb];
                    v8h x0 = *(const v8h*)r0;
                    v8h x1 = *(const v8h*)(r0 + 16);
                    #pragma unroll
                    for (int t = 0; t < 8; ++t) { a0[t] = x0[t]; a0[8 + t] = x1[t]; }
                }
                {
                    const _Float16* r0 = &ldsK[buf][16 + lh][32 * f + 8 * hb];
                    v8h x0 = *(const v8h*)r0;
                    v8h x1 = *(const v8h*)(r0 + 16);
                    #pragma unroll
                    for (int t = 0; t < 8; ++t) { a1[t] = x0[t]; a1[8 + t] = x1[t]; }
                }
                c1 = __builtin_amdgcn_wmma_f32_16x16x32_f16(false, a0, false, bq[f], (short)0, c1, false, false);
                c2 = __builtin_amdgcn_wmma_f32_16x16x32_f16(false, a1, false, bq[f], (short)0, c2, false, false);
            }

            // ---- scale (log2 domain); mask only in the diagonal-straddling blocks ----
            #pragma unroll
            for (int r = 0; r < 8; ++r) { c1[r] *= kScale2; c2[r] *= kScale2; }
            if (i >= numFull) {                      // wave-uniform branch
                const int rel1 = qrow - kb - 8 * hb; // mask c1[r] iff r > rel1
                const int rel2 = rel1 - 16;          // mask c2[r] iff r > rel2
                #pragma unroll
                for (int r = 0; r < 8; ++r) {
                    if (r > rel1) c1[r] = kMask2;
                    if (r > rel2) c2[r] = kMask2;
                }
            }

            // ---- online softmax (row split across lanes q / q+16, one xor-16 combine) ----
            float mb = -3.0e38f;
            #pragma unroll
            for (int r = 0; r < 8; ++r) mb = fmaxf(mb, fmaxf(c1[r], c2[r]));
            mb = fmaxf(mb, __shfl_xor(mb, 16, 32));
            const float mn    = fmaxf(m, mb);
            const float alpha = fexp2(m - mn);
            float ps = 0.0f;
            #pragma unroll
            for (int r = 0; r < 8; ++r) {
                c1[r] = fexp2(c1[r] - mn);
                c2[r] = fexp2(c2[r] - mn);
                ps += c1[r] + c2[r];
            }
            ps += __shfl_xor(ps, 16, 32);
            l = l * alpha + ps;
            m = mn;
            #pragma unroll
            for (int t = 0; t < 8; ++t) o[t] *= alpha;   // packs to v_pk_mul_f32

            // ---- pack P^T into B-operand: lane q holds keys kb+0..15 (lo) / kb+16..31 (hi) ----
            v16h bp;
            #pragma unroll
            for (int r = 0; r < 8; ++r) {
                const float t1 = __shfl_xor(c1[r], 16, 32);
                const float t2 = __shfl_xor(c2[r], 16, 32);
                bp[r]     = (_Float16)(hb ? t2 : c1[r]);   // K = r    (lo) / 16+r (hi)
                bp[8 + r] = (_Float16)(hb ? c2[r] : t1);   // K = 8+r  (lo) / 24+r (hi)
            }

            // ---- GEMM 2: O^T += V^T_tile(16x32) * P^T(32x16), 8 vdim tiles ----
            #pragma unroll
            for (int t = 0; t < 8; ++t) {
                v16h av;
                const _Float16* vr = &ldsVt[buf][16 * t + lh][16 * hb];
                v8h y0 = *(const v8h*)vr;
                v8h y1 = *(const v8h*)(vr + 8);
                #pragma unroll
                for (int u = 0; u < 8; ++u) { av[u] = y0[u]; av[8 + u] = y1[u]; }
                o[t] = __builtin_amdgcn_wmma_f32_16x16x32_f16(false, av, false, bp, (short)0, o[t], false, false);
            }
        }

        // D: stage block i+1 into the other buffer (reads of it were fenced by barrier B)
        if (i + 1 < nblk) storeBlk(buf ^ 1);
    }

    // ---- epilogue: O[qrow][vdim] = O^T / l ; lane q holds vdims 16t+8*hb+0..7 per tile ----
    const float inv = 1.0f / l;
    float* op = Out + ((size_t)b * Nc + qrow) * Vc;
    #pragma unroll
    for (int t = 0; t < 8; ++t) {
        const int vbase = 16 * t + 8 * hb;
        v4f lo4, hi4;
        #pragma unroll
        for (int r = 0; r < 4; ++r) { lo4[r] = o[t][r] * inv; hi4[r] = o[t][4 + r] * inv; }
        *(v4f*)(op + vbase)     = lo4;
        *(v4f*)(op + vbase + 4) = hi4;
    }
}

extern "C" void kernel_launch(void* const* d_in, const int* in_sizes, int n_in,
                              void* d_out, int out_size, void* d_ws, size_t ws_size,
                              hipStream_t stream) {
    (void)in_sizes; (void)n_in; (void)out_size; (void)d_ws; (void)ws_size;
    const float* Q   = (const float*)d_in[0];
    const float* K   = (const float*)d_in[1];
    const float* Vv  = (const float*)d_in[2];
    const int*   kpl = (const int*)d_in[3];
    const int*   afl = (const int*)d_in[4];
    float* Out = (float*)d_out;

    dim3 grid(Bc * (Nc / QT));   // 256 blocks: one per (batch, 128-row Q tile)
    dim3 block(256);             // 8 waves
    fa2_wmma_gfx1250<<<grid, block, 0, stream>>>(Q, K, Vv, kpl, afl, Out);
}
